// GCN_27427661152789
// MI455X (gfx1250) — matrix operand order repaired and checked by reading, not compiled, not run on hardware
//
#include <hip/hip_runtime.h>
#include <hip/hip_bf16.h>

#define N_NODES 100000
#define N_EDGES 1600000
#define NFEAT   512
#define NHID    64
#define NCLASS  40

typedef __bf16  v16bf __attribute__((ext_vector_type(16)));
typedef float   v8f   __attribute__((ext_vector_type(8)));

union ABf16 {
    __bf16         h[16];
    unsigned short s[16];
    uint4          q[2];
    v16bf          v;
};

// ---------------------------------------------------------------------------
// Convert W1 [512][64] f32 -> W1T [64][512] bf16 (column-major for B loads)
__global__ void gcn_cvt_w1(const float* __restrict__ W1,
                           __bf16* __restrict__ W1T) {
    int i = blockIdx.x * blockDim.x + threadIdx.x;
    if (i >= NFEAT * NHID) return;
    int k = i >> 6;          // 0..511
    int c = i & 63;          // 0..63
    W1T[c * NFEAT + k] = (__bf16)W1[i];
}

// Convert W10 [64][40] f32 -> W10T [48][64] bf16, cols 40..47 zero-padded
__global__ void gcn_cvt_w10(const float* __restrict__ W10,
                            __bf16* __restrict__ W10T) {
    int i = blockIdx.x * blockDim.x + threadIdx.x;
    if (i >= 48 * 64) return;
    int c = i >> 6;          // 0..47
    int k = i & 63;          // 0..63
    float v = (c < NCLASS) ? W10[k * NCLASS + c] : 0.0f;
    W10T[c * 64 + k] = (__bf16)v;
}

// ---------------------------------------------------------------------------
// Load one 16x32 bf16 A-tile for this lane from an f32 row (optional ReLU).
template <bool RELU>
__device__ __forceinline__ v16bf load_a_tile(const float* __restrict__ rowp,
                                             int ks, int half) {
    const float4* pa = reinterpret_cast<const float4*>(rowp + ks * 32 + half * 16);
    const float4* pb = reinterpret_cast<const float4*>(rowp + ks * 32 + half * 16 + 8);
    float4 a0 = pa[0], a1 = pa[1];
    float4 a2 = pb[0], a3 = pb[1];
    float f[16] = {a0.x, a0.y, a0.z, a0.w, a1.x, a1.y, a1.z, a1.w,
                   a2.x, a2.y, a2.z, a2.w, a3.x, a3.y, a3.z, a3.w};
    ABf16 A;
    #pragma unroll
    for (int i = 0; i < 16; ++i) {
        float v = RELU ? fmaxf(f[i], 0.0f) : f[i];
        A.h[i] = (__bf16)v;     // native fptrunc -> v_cvt (RNE)
    }
    return A.v;
}

// ---------------------------------------------------------------------------
// GEMM1: support1[100000][64] = x[100000][512] @ W1[512][64]   (bf16 WMMA)
// grid 3125 blocks x 128 threads; wave w computes cols [16w,16w+16) of a
// 32-row stripe (two 16x16 WMMA row-tiles sharing the B tile).
__global__ __launch_bounds__(128) void gcn_gemm1(
        const float* __restrict__ x,
        const __bf16* __restrict__ W1T,   // [64][512] bf16
        float* __restrict__ support1) {
    const int tid  = threadIdx.x;
    const int wave = tid >> 5;
    const int lane = tid & 31;
    const int half = lane >> 4;       // 0/1
    const int l16  = lane & 15;
    const int rowBase = blockIdx.x * 32;
    const int col  = wave * 16 + l16;         // B-matrix N index for this lane

    const float* xrow0 = x + (size_t)(rowBase + l16) * NFEAT;
    const float* xrow1 = x + (size_t)(rowBase + 16 + l16) * NFEAT;
    const __bf16* bcol = W1T + (size_t)col * NFEAT;

    v8f acc0 = {0.f, 0.f, 0.f, 0.f, 0.f, 0.f, 0.f, 0.f};
    v8f acc1 = {0.f, 0.f, 0.f, 0.f, 0.f, 0.f, 0.f, 0.f};

    for (int ks = 0; ks < NFEAT / 32; ++ks) {
        // B: 32x16 bf16 tile, lane holds K = half*16 + e at column `col`
        const uint4* bp = reinterpret_cast<const uint4*>(bcol + ks * 32 + half * 16);
        ABf16 B;
        B.q[0] = bp[0];
        B.q[1] = bp[1];

        v16bf A0 = load_a_tile<false>(xrow0, ks, half);
        v16bf A1 = load_a_tile<false>(xrow1, ks, half);

        acc0 = __builtin_amdgcn_wmma_f32_16x16x32_bf16(
                false, A0, false, B.v, (short)0, acc0, false, false);
        acc1 = __builtin_amdgcn_wmma_f32_16x16x32_bf16(
                false, A1, false, B.v, (short)0, acc1, false, false);
    }

    // D layout: lane holds D[half*8 + r][l16] in acc[r]
    #pragma unroll
    for (int r = 0; r < 8; ++r) {
        int m0 = rowBase + half * 8 + r;
        support1[(size_t)m0 * NHID + wave * 16 + l16] = acc0[r];
    }
    #pragma unroll
    for (int r = 0; r < 8; ++r) {
        int m1 = rowBase + 16 + half * 8 + r;
        support1[(size_t)m1 * NHID + wave * 16 + l16] = acc1[r];
    }
}

// ---------------------------------------------------------------------------
// GEMM2: support2[100000][40] = relu(h)[100000][64] @ W10[64][40]
// grid 1563 blocks x 128 threads; each wave owns one 16-row tile, 3 col tiles.
__global__ __launch_bounds__(128) void gcn_gemm2(
        const float* __restrict__ h,
        const __bf16* __restrict__ W10T,  // [48][64] bf16
        float* __restrict__ support2) {
    const int tid  = threadIdx.x;
    const int wave = tid >> 5;
    const int lane = tid & 31;
    const int half = lane >> 4;
    const int l16  = lane & 15;
    const int rowTile = blockIdx.x * 4 + wave;
    if (rowTile >= N_NODES / 16) return;
    const int row = rowTile * 16 + l16;
    const float* hrow = h + (size_t)row * NHID;

    v8f acc[3];
    #pragma unroll
    for (int t = 0; t < 3; ++t)
        acc[t] = (v8f){0.f, 0.f, 0.f, 0.f, 0.f, 0.f, 0.f, 0.f};

    #pragma unroll
    for (int ks = 0; ks < 2; ++ks) {
        v16bf A = load_a_tile<true>(hrow, ks, half);   // ReLU fused

        #pragma unroll
        for (int ct = 0; ct < 3; ++ct) {
            const int col = ct * 16 + l16;   // padded to 48
            const uint4* bp = reinterpret_cast<const uint4*>(
                    W10T + (size_t)col * 64 + ks * 32 + half * 16);
            ABf16 B;
            B.q[0] = bp[0];
            B.q[1] = bp[1];
            acc[ct] = __builtin_amdgcn_wmma_f32_16x16x32_bf16(
                    false, A, false, B.v, (short)0, acc[ct], false, false);
        }
    }

    #pragma unroll
    for (int ct = 0; ct < 3; ++ct) {
        const int col = ct * 16 + l16;
        if (col < NCLASS) {
            #pragma unroll
            for (int r = 0; r < 8; ++r) {
                int m = rowTile * 16 + half * 8 + r;
                support2[(size_t)m * NCLASS + col] = acc[ct][r];
            }
        }
    }
}

// ---------------------------------------------------------------------------
// buf[n][j] = bias[j]
__global__ void gcn_init_bias(float* __restrict__ buf,
                              const float* __restrict__ bias,
                              int ncols, int total) {
    int i = blockIdx.x * blockDim.x + threadIdx.x;
    if (i < total) buf[i] = bias[i % ncols];
}

// COO scatter: dst[row[e]] += vals[e] * src[col[e]], chunks of 4 floats.
__global__ void gcn_spmm_scatter(const float* __restrict__ vals,
                                 const int* __restrict__ erow,
                                 const int* __restrict__ ecol,
                                 const float* __restrict__ src,
                                 float* __restrict__ dst,
                                 int chunks, int ncols) {
    int gid = blockIdx.x * blockDim.x + threadIdx.x;
    int e = gid / chunks;
    int q = gid - e * chunks;
    if (e >= N_EDGES) return;
    int r = erow[e];
    int c = ecol[e];
    float v = vals[e];
    const float4 s = *reinterpret_cast<const float4*>(src + (size_t)c * ncols + q * 4);
    float* d = dst + (size_t)r * ncols + q * 4;
    atomicAdd(d + 0, v * s.x);
    atomicAdd(d + 1, v * s.y);
    atomicAdd(d + 2, v * s.z);
    atomicAdd(d + 3, v * s.w);
}

// ---------------------------------------------------------------------------
// In-place log_softmax over 40 classes, one wave per row.
__global__ __launch_bounds__(128) void gcn_log_softmax(float* __restrict__ out) {
    const int wave = threadIdx.x >> 5;
    const int lane = threadIdx.x & 31;
    const int row  = blockIdx.x * 4 + wave;
    if (row >= N_NODES) return;
    float* p = out + (size_t)row * NCLASS;

    float v0 = p[lane];                                   // lanes 0..31 valid
    float v1 = (lane < NCLASS - 32) ? p[32 + lane] : -INFINITY;

    float m = fmaxf(v0, v1);
    #pragma unroll
    for (int o = 16; o >= 1; o >>= 1)
        m = fmaxf(m, __shfl_xor(m, o, 32));

    float s = __expf(v0 - m) + ((lane < NCLASS - 32) ? __expf(v1 - m) : 0.f);
    #pragma unroll
    for (int o = 16; o >= 1; o >>= 1)
        s += __shfl_xor(s, o, 32);

    float ls = __logf(s);
    p[lane] = v0 - m - ls;
    if (lane < NCLASS - 32) p[32 + lane] = v1 - m - ls;
}

// ---------------------------------------------------------------------------
extern "C" void kernel_launch(void* const* d_in, const int* in_sizes, int n_in,
                              void* d_out, int out_size, void* d_ws, size_t ws_size,
                              hipStream_t stream) {
    const float* x    = (const float*)d_in[0];
    const float* adj  = (const float*)d_in[1];
    const float* W1   = (const float*)d_in[2];
    const float* b1   = (const float*)d_in[3];
    const float* W10  = (const float*)d_in[4];
    const float* b10  = (const float*)d_in[5];
    const int*   erow = (const int*)d_in[6];
    const int*   ecol = (const int*)d_in[7];
    float* out = (float*)d_out;

    // workspace layout
    float* support1 = (float*)d_ws;                                  // N*64
    float* h        = support1 + (size_t)N_NODES * NHID;             // N*64
    float* support2 = h        + (size_t)N_NODES * NHID;             // N*40
    __bf16* W1T  = (__bf16*)(support2 + (size_t)N_NODES * NCLASS);
    __bf16* W10T = W1T + (size_t)NHID * NFEAT;

    // 0) weight conversion (bf16, transposed)
    gcn_cvt_w1 <<<(NFEAT * NHID + 255) / 256, 256, 0, stream>>>(W1, W1T);
    gcn_cvt_w10<<<(48 * 64 + 255) / 256,       256, 0, stream>>>(W10, W10T);

    // 1) support1 = x @ W1  (WMMA bf16)
    gcn_gemm1<<<N_NODES / 32, 128, 0, stream>>>(x, W1T, support1);

    // 2) h = b1 + spmm(adj, support1); ReLU fused into gemm2 load
    gcn_init_bias<<<(N_NODES * NHID + 255) / 256, 256, 0, stream>>>(
            h, b1, NHID, N_NODES * NHID);
    gcn_spmm_scatter<<<(N_EDGES * 16 + 255) / 256, 256, 0, stream>>>(
            adj, erow, ecol, support1, h, 16, NHID);

    // 3) support2 = relu(h) @ W10  (WMMA bf16)
    gcn_gemm2<<<(N_NODES / 16 + 3) / 4, 128, 0, stream>>>(h, W10T, support2);

    // 4) out = b10 + spmm(adj, support2)
    gcn_init_bias<<<(N_NODES * NCLASS + 255) / 256, 256, 0, stream>>>(
            out, b10, NCLASS, N_NODES * NCLASS);
    gcn_spmm_scatter<<<(N_EDGES * 10 + 255) / 256, 256, 0, stream>>>(
            adj, erow, ecol, support2, out, 10, NCLASS);

    // 5) log_softmax in place
    gcn_log_softmax<<<(N_NODES + 3) / 4, 128, 0, stream>>>(out);
}
